// CachedMultiheadAttention_69002944578184
// MI455X (gfx1250) — matrix operand order repaired and checked by reading, not compiled, not run on hardware
//
#include <hip/hip_runtime.h>

// MI455X / gfx1250, wave32. All matrix math via v_wmma_f32_16x16x32_bf16.
// ~86 GFLOP over ~84MB I/O -> compute bound -> bf16 WMMA everywhere.
// GEMM waves compute 32x64 tiles (2 A-frags reuse 4 B-frags -> 8 WMMA / k-step).

typedef __attribute__((ext_vector_type(16))) __bf16 v16bf;
typedef __attribute__((ext_vector_type(8)))  __bf16 v8bf;
typedef __attribute__((ext_vector_type(8)))  float  v8f;

#define S_LEN   2048
#define EMB     1024
#define NH      16
#define HD      64
#define WIN_SZ  512
#define BATCH   4
#define NROW    (BATCH * S_LEN)   /* 8192 rows of the token matrix */

__device__ __forceinline__ unsigned short f2bf(float x) {
  unsigned u = __float_as_uint(x);
  u += 0x7FFFu + ((u >> 16) & 1u);      // round-to-nearest-even
  return (unsigned short)(u >> 16);
}
__device__ __forceinline__ __bf16 us2bf(unsigned short u) {
  union { unsigned short u; __bf16 b; } c; c.u = u; return c.b;
}

__device__ __forceinline__ v8f wmma_bf16(v16bf a, v16bf b, v8f c) {
  // (neg_a, A, neg_b, B, c_mod, C, reuse_a, reuse_b)
  return __builtin_amdgcn_wmma_f32_16x16x32_bf16(false, a, false, b, (short)0, c,
                                                 false, false);
}

// A-matrix 16x32 bf16 fragment (ISA 7.12.2): lane m = L%16, hi = L/16.
// element i <-> K = (i<8 ? i : i+8) + hi*8  => two contiguous 8-element runs.
// Caller passes pointer at row*K + k0 + hi*8.
__device__ __forceinline__ v16bf load_a_frag(const unsigned short* p) {
  v8bf lo = *(const v8bf*)(const void*)(p);
  v8bf hh = *(const v8bf*)(const void*)(p + 16);
  v16bf r;
#pragma unroll
  for (int i = 0; i < 8; ++i) { r[i] = lo[i]; r[8 + i] = hh[i]; }
  return r;
}
// B-matrix 32x16 bf16 fragment: lane n = L%16, element i <-> K = (i<8 ? i : i+8) + hi*8.
// Caller passes pointer at nrow*K + k0 + hi*8 (16B aligned, two 8-element runs).
__device__ __forceinline__ v16bf load_b_frag(const unsigned short* p) {
  v8bf lo = *(const v8bf*)(const void*)(p);
  v8bf hh = *(const v8bf*)(const void*)(p + 16);
  v16bf r;
#pragma unroll
  for (int i = 0; i < 8; ++i) { r[i] = lo[i]; r[8 + i] = hh[i]; }
  return r;
}

// ---------------------------------------------------------------- converts
__global__ void cvt_f32_to_bf16(const float* __restrict__ in,
                                unsigned short* __restrict__ out, int n) {
  int i = blockIdx.x * blockDim.x + threadIdx.x;
  int stride = gridDim.x * blockDim.x;
  for (; i < n; i += stride) out[i] = f2bf(in[i]);
}

// ------------------------------------------------------------- QKV GEMM
// C[8192,3072] = Xb[8192,1024] x Win^T ; wave computes 32 rows x 64 cols.
// 64-col group == exactly one head slice => simple scatter into q/k/vT.
__global__ __launch_bounds__(32)
void qkv_gemm(const unsigned short* __restrict__ xb,
              const unsigned short* __restrict__ wib,
              unsigned short* __restrict__ qb,
              unsigned short* __restrict__ kb,
              unsigned short* __restrict__ vtb) {
  const int lane = threadIdx.x & 31;
  const int m = lane & 15, hi = lane >> 4;
  const int row0 = blockIdx.x * 32;
  const int n0   = blockIdx.y * 64;

  v8f acc[2][4];
#pragma unroll
  for (int g = 0; g < 2; ++g)
#pragma unroll
    for (int t = 0; t < 4; ++t)
#pragma unroll
      for (int r = 0; r < 8; ++r) acc[g][t][r] = 0.0f;

  const unsigned short* arow0 = xb + (size_t)(row0 + m) * EMB + hi * 8;
  const unsigned short* arow1 = xb + (size_t)(row0 + 16 + m) * EMB + hi * 8;

  for (int k0 = 0; k0 < EMB; k0 += 32) {
    if (k0 + 32 < EMB) {                         // stream-prefetch next A chunk
      __builtin_prefetch(arow0 + k0 + 32, 0, 1);
      __builtin_prefetch(arow1 + k0 + 32, 0, 1);
    }
    v16bf a0 = load_a_frag(arow0 + k0);
    v16bf a1 = load_a_frag(arow1 + k0);
#pragma unroll
    for (int t = 0; t < 4; ++t) {
      v16bf b = load_b_frag(wib + (size_t)(n0 + t * 16 + m) * EMB + k0 + hi * 8);
      acc[0][t] = wmma_bf16(a0, b, acc[0][t]);
      acc[1][t] = wmma_bf16(a1, b, acc[1][t]);
    }
  }
  const int seg = n0 >> 10;              // 0:q 1:k 2:v
  const int h   = (n0 & 1023) >> 6;
#pragma unroll
  for (int g = 0; g < 2; ++g)
#pragma unroll
    for (int t = 0; t < 4; ++t) {
      const int d = t * 16 + m;          // n0 is 64-aligned within the segment
#pragma unroll
      for (int r = 0; r < 8; ++r) {
        const int row = row0 + g * 16 + r + hi * 8;   // row = b*S + s
        const int bb = row >> 11, s = row & (S_LEN - 1);
        const int bh = bb * NH + h;
        const unsigned short v = f2bf(acc[g][t][r]);
        if (seg == 0)      qb [((size_t)bh * S_LEN + s) * HD + d] = v;
        else if (seg == 1) kb [((size_t)bh * S_LEN + s) * HD + d] = v;
        else               vtb[((size_t)bh * HD + d) * S_LEN + s] = v;  // transposed
      }
    }
}

// ------------------------------------------------ sliding-window attention
// One wave per (b,h, 16-query tile). Flash-style online softmax; ALiBi bias.
__global__ __launch_bounds__(32)
void swa_attn(const unsigned short* __restrict__ qb,
              const unsigned short* __restrict__ kb,
              const unsigned short* __restrict__ vtb,
              unsigned short* __restrict__ ob) {
  __shared__ alignas(32) unsigned short sP[16 * 32];   // P tile bounce (1KB)

  const int lane = threadIdx.x & 31;
  const int m = lane & 15, hi = lane >> 4;
  const int idx = blockIdx.x;                      // B*H*(S/16) waves
  const int bh = idx >> 7;                         // S/16 == 128
  const int q0 = (idx & 127) << 4;
  const int b  = bh >> 4, h = bh & 15;

  const float scale = 0.125f;                      // 1/sqrt(64)
  const float slope = exp2f(-(float)(h + 1) * 0.5f);   // exp2(-(h+1)*8/H)

  v16bf aq[2];
#pragma unroll
  for (int c = 0; c < 2; ++c)
    aq[c] = load_a_frag(qb + ((size_t)bh * S_LEN + q0 + m) * HD + c * 32 + hi * 8);

  float mrow[8], lrow[8];
  v8f acc[4];
#pragma unroll
  for (int r = 0; r < 8; ++r) { mrow[r] = -3.0e38f; lrow[r] = 0.0f; }
#pragma unroll
  for (int t = 0; t < 4; ++t)
#pragma unroll
    for (int r = 0; r < 8; ++r) acc[t][r] = 0.0f;

  int kv_start = q0 - WIN_SZ; if (kv_start < 0) kv_start = 0;

  for (int kv0 = kv_start; kv0 < q0 + 16; kv0 += 32) {
    // ---- scores: two 16x16 C tiles covering 32 keys
    v8f sc[2];
#pragma unroll
    for (int t = 0; t < 2; ++t) {
      int kv = kv0 + t * 16 + m;
      if (kv > S_LEN - 1) kv = S_LEN - 1;          // clamped; masked below
      v8f c = {0,0,0,0,0,0,0,0};
#pragma unroll
      for (int cc = 0; cc < 2; ++cc) {
        v16bf bk = load_b_frag(kb + ((size_t)bh * S_LEN + kv) * HD + cc * 32 + hi * 8);
        c = wmma_bf16(aq[cc], bk, c);
      }
      sc[t] = c;
    }
    // ---- bias + mask + online softmax (row lives across 16 lanes of a half)
#pragma unroll
    for (int r = 0; r < 8; ++r) {
      const int qi = q0 + r + hi * 8;
      const int d0 = qi - (kv0 + m);
      const int d1 = qi - (kv0 + 16 + m);
      float x0 = (d0 >= 0 && d0 <= WIN_SZ) ? sc[0][r] * scale + slope * (float)d0 : -3.0e38f;
      float x1 = (d1 >= 0 && d1 <= WIN_SZ) ? sc[1][r] * scale + slope * (float)d1 : -3.0e38f;
      float mx = fmaxf(x0, x1);
#pragma unroll
      for (int o = 1; o < 16; o <<= 1) mx = fmaxf(mx, __shfl_xor(mx, o, 32));
      const float mnew  = fmaxf(mrow[r], mx);
      const float alpha = __expf(mrow[r] - mnew);
      const float p0 = __expf(x0 - mnew);
      const float p1 = __expf(x1 - mnew);
      float rs = p0 + p1;
#pragma unroll
      for (int o = 1; o < 16; o <<= 1) rs += __shfl_xor(rs, o, 32);
      lrow[r] = lrow[r] * alpha + rs;
      mrow[r] = mnew;
#pragma unroll
      for (int dt = 0; dt < 4; ++dt) acc[dt][r] *= alpha;
      sP[(r + hi * 8) * 32 + m]      = f2bf(p0);   // P in row-major 16x32
      sP[(r + hi * 8) * 32 + 16 + m] = f2bf(p1);
    }
    asm volatile("s_wait_dscnt 0" ::: "memory");   // wave-internal LDS RAW fence
    // ---- P as A-fragment: two 16B vector LDS loads (ds_load_b128)
    v16bf pa;
    {
      v8bf lo = *(const v8bf*)(const void*)&sP[m * 32 + hi * 8];
      v8bf hh = *(const v8bf*)(const void*)&sP[m * 32 + 16 + hi * 8];
#pragma unroll
      for (int i = 0; i < 8; ++i) { pa[i] = lo[i]; pa[8 + i] = hh[i]; }
    }
    // ---- PV: 4 output d-tiles, V pre-transposed so B-frags are contiguous
#pragma unroll
    for (int dt = 0; dt < 4; ++dt) {
      v16bf bv = load_b_frag(vtb + ((size_t)bh * HD + dt * 16 + m) * S_LEN + kv0 + hi * 8);
      acc[dt] = wmma_bf16(pa, bv, acc[dt]);
    }
    asm volatile("s_wait_dscnt 0" ::: "memory");   // keep next-iter stores after loads
  }
  // ---- normalize + merge heads into [B,S,E] bf16
#pragma unroll
  for (int dt = 0; dt < 4; ++dt)
#pragma unroll
    for (int r = 0; r < 8; ++r) {
      const int s = q0 + r + hi * 8;
      const float v = acc[dt][r] / lrow[r];
      ob[((size_t)b * S_LEN + s) * EMB + h * HD + dt * 16 + m] = f2bf(v);
    }
}

// ------------------------------------------------------------ out-proj GEMM
__global__ __launch_bounds__(32)
void out_gemm(const unsigned short* __restrict__ ob,
              const unsigned short* __restrict__ wob,
              float* __restrict__ out) {
  const int lane = threadIdx.x & 31;
  const int m = lane & 15, hi = lane >> 4;
  const int row0 = blockIdx.x * 32;
  const int n0   = blockIdx.y * 64;

  v8f acc[2][4];
#pragma unroll
  for (int g = 0; g < 2; ++g)
#pragma unroll
    for (int t = 0; t < 4; ++t)
#pragma unroll
      for (int r = 0; r < 8; ++r) acc[g][t][r] = 0.0f;

  const unsigned short* arow0 = ob + (size_t)(row0 + m) * EMB + hi * 8;
  const unsigned short* arow1 = ob + (size_t)(row0 + 16 + m) * EMB + hi * 8;

  for (int k0 = 0; k0 < EMB; k0 += 32) {
    if (k0 + 32 < EMB) {
      __builtin_prefetch(arow0 + k0 + 32, 0, 1);
      __builtin_prefetch(arow1 + k0 + 32, 0, 1);
    }
    v16bf a0 = load_a_frag(arow0 + k0);
    v16bf a1 = load_a_frag(arow1 + k0);
#pragma unroll
    for (int t = 0; t < 4; ++t) {
      v16bf b = load_b_frag(wob + (size_t)(n0 + t * 16 + m) * EMB + k0 + hi * 8);
      acc[0][t] = wmma_bf16(a0, b, acc[0][t]);
      acc[1][t] = wmma_bf16(a1, b, acc[1][t]);
    }
  }
#pragma unroll
  for (int g = 0; g < 2; ++g)
#pragma unroll
    for (int t = 0; t < 4; ++t)
#pragma unroll
      for (int r = 0; r < 8; ++r)
        out[(size_t)(row0 + g * 16 + r + hi * 8) * EMB + n0 + t * 16 + m] = acc[g][t][r];
}

// --------------------------------------------------------------- launcher
extern "C" void kernel_launch(void* const* d_in, const int* in_sizes, int n_in,
                              void* d_out, int out_size, void* d_ws, size_t ws_size,
                              hipStream_t stream) {
  const float* x     = (const float*)d_in[0];   // [B,S,E]
  const float* w_in  = (const float*)d_in[1];   // [3E,E]
  const float* w_out = (const float*)d_in[2];   // [E,E]
  float* out = (float*)d_out;

  const size_t NX  = (size_t)NROW * EMB;        // 8,388,608
  const size_t NWI = (size_t)3 * EMB * EMB;     // 3,145,728
  const size_t NWO = (size_t)EMB * EMB;         // 1,048,576

  char* ws = (char*)d_ws;
  size_t off = 0;
  auto carve = [&](size_t bytes) -> unsigned short* {
    unsigned short* p = (unsigned short*)(ws + off);
    off += (bytes + 255) & ~(size_t)255;
    return p;
  };
  unsigned short* xb  = carve(NX  * 2);   // x bf16; reused as attn-out later
  unsigned short* wib = carve(NWI * 2);
  unsigned short* wob = carve(NWO * 2);
  unsigned short* qb  = carve(NX * 2);    // [B,H,S,D]
  unsigned short* kb  = carve(NX * 2);    // [B,H,S,D]
  unsigned short* vtb = carve(NX * 2 + 256); // [B,H,D,S] (+pad for clamped tail reads)
  unsigned short* obuf = xb;              // alias: xb dead after qkv_gemm

  cvt_f32_to_bf16<<<4096, 256, 0, stream>>>(x,     xb,  (int)NX);
  cvt_f32_to_bf16<<<2048, 256, 0, stream>>>(w_in,  wib, (int)NWI);
  cvt_f32_to_bf16<<<1024, 256, 0, stream>>>(w_out, wob, (int)NWO);

  dim3 g1(NROW / 32, (3 * EMB) / 64);     // 256 x 48
  qkv_gemm<<<g1, 32, 0, stream>>>(xb, wib, qb, kb, vtb);

  swa_attn<<<BATCH * NH * (S_LEN / 16), 32, 0, stream>>>(qb, kb, vtb, obuf);

  dim3 g2(NROW / 32, EMB / 64);           // 256 x 16
  out_gemm<<<g2, 32, 0, stream>>>(obuf, wob, out);
}
